// GaussianImage_Cholesky_36524401885824
// MI455X (gfx1250) — matrix operand order repaired and checked by reading, not compiled, not run on hardware
//
#include <hip/hip_runtime.h>
#include <hip/hip_bf16.h>

typedef __attribute__((ext_vector_type(16))) _Float16 v16h;
typedef __attribute__((ext_vector_type(8)))  float    v8f;

#define NGAUSS 2048
#define IMG_W  256
#define IMG_H  256
#define NPIX   (IMG_W * IMG_H)
#define NBLK   (NGAUSS / 32)   // 64 K-blocks of 32 gaussians
#define LOG2E  1.4426950408889634f

__device__ __forceinline__ float fast_sigmoid(float x) {
    return 1.0f / (1.0f + __expf(-x));
}

// ---------------------------------------------------------------------------
// Kernel 1: per-gaussian parameter precompute (time polynomial, tanh means,
// Cholesky -> inverse covariance, sigmoid opacity). One thread per gaussian.
// Quadratic coefficients are stored pre-negated and pre-scaled by log2(e) so
// the render inner loop is two FMAs feeding a bare v_exp_f32:
//   packA = (nhA, nB, nhC, cx) where nhA = -0.5*A*log2e, nB = -B*log2e,
//                                    nhC = -0.5*C*log2e
//   packB = (cy, opac)
// ---------------------------------------------------------------------------
__global__ void gi_params_kernel(const float* __restrict__ xyz,     // (3, N, 2)
                                 const float* __restrict__ chol,    // (3, N, 3)
                                 const float* __restrict__ opacity, // (N, 1)
                                 const int*   __restrict__ frame_index,
                                 float4* __restrict__ packA,
                                 float2* __restrict__ packB)
{
    int n = blockIdx.x * blockDim.x + threadIdx.x;
    if (n >= NGAUSS) return;

    float t  = (float)frame_index[0] * (1.0f / 7.0f);  // linspace(0,1,T=8)
    float t2 = t * t;

    float mx = tanhf(xyz[0 * NGAUSS * 2 + n * 2 + 0]
             + t  * xyz[1 * NGAUSS * 2 + n * 2 + 0]
             + t2 * xyz[2 * NGAUSS * 2 + n * 2 + 0]);
    float my = tanhf(xyz[0 * NGAUSS * 2 + n * 2 + 1]
             + t  * xyz[1 * NGAUSS * 2 + n * 2 + 1]
             + t2 * xyz[2 * NGAUSS * 2 + n * 2 + 1]);

    float l1 = chol[0 * NGAUSS * 3 + n * 3 + 0]
             + t  * chol[1 * NGAUSS * 3 + n * 3 + 0]
             + t2 * chol[2 * NGAUSS * 3 + n * 3 + 0] + 0.5f;
    float l2 = chol[0 * NGAUSS * 3 + n * 3 + 1]
             + t  * chol[1 * NGAUSS * 3 + n * 3 + 1]
             + t2 * chol[2 * NGAUSS * 3 + n * 3 + 1];
    float l3 = chol[0 * NGAUSS * 3 + n * 3 + 2]
             + t  * chol[1 * NGAUSS * 3 + n * 3 + 2]
             + t2 * chol[2 * NGAUSS * 3 + n * 3 + 2] + 0.5f;

    float s11 = l1 * l1;
    float s12 = l1 * l2;
    float s22 = l2 * l2 + l3 * l3;
    float det = s11 * s22 - s12 * s12;
    float inv = 1.0f / det;
    float Aq  =  s22 * inv;
    float Bq  = -s12 * inv;
    float Cq  =  s11 * inv;

    // pre-negate and pre-scale: sigma*log2e folded into coefficients
    float nhA = -0.5f * LOG2E * Aq;
    float nB  = -LOG2E * Bq;
    float nhC = -0.5f * LOG2E * Cq;

    float cx = 0.5f * (float)IMG_W * (mx + 1.0f);
    float cy = 0.5f * (float)IMG_H * (my + 1.0f);
    float op = fast_sigmoid(opacity[n]);

    packA[n] = make_float4(nhA, nB, nhC, cx);
    packB[n] = make_float2(cy, op);
}

// ---------------------------------------------------------------------------
// Kernel 2: build the WMMA A-matrix (colors^T, 16x32 per K-block) directly in
// the CDNA5 16-bit A-matrix VGPR layout so the render kernel can stream it
// with one 32-byte load per lane per K-step.
// A layout (16-bit, 16x32): lane L -> M = L&15; lanes 0-15 carry K={0..7,16..23},
// lanes 16-31 carry K={8..15,24..31}; element e -> K = base + e + (e&8).
// Rows M>=3 are zero (only 3 color channels used).
// ---------------------------------------------------------------------------
__global__ void gi_amat_kernel(const float* __restrict__ features, // (N, 3)
                               _Float16* __restrict__ amat)        // NBLK*32*16 halfs
{
    int tid = blockIdx.x * blockDim.x + threadIdx.x;   // 0 .. NBLK*32-1
    if (tid >= NBLK * 32) return;
    int b = tid >> 5;
    int l = tid & 31;
    int m = l & 15;
    int kbase = b * 32 + ((l & 16) ? 8 : 0);

#pragma unroll
    for (int e = 0; e < 16; ++e) {
        int k = kbase + e + (e & 8);
        float v = (m < 3) ? fast_sigmoid(features[k * 3 + m]) : 0.0f;
        amat[tid * 16 + e] = (_Float16)v;
    }
}

// ---------------------------------------------------------------------------
// Kernel 3: render. One workgroup = 8 waves = 256 contiguous pixels.
// Each wave owns 32 pixels (two 16-pixel WMMA column groups) and loops over
// all 64 K-blocks of 32 gaussians. Per K-step each lane evaluates 16 alphas
// (2 FMAs -> v_exp_f32 -> min -> packed f16 cvt) for each of its 2 pixel
// groups, then two v_wmma_f32_16x16x32_f16 accumulate RGB, co-executing with
// the next step's VALU work.
// Per-gaussian params live in 48 KB of LDS (broadcast ds_load_b128/b64).
// ---------------------------------------------------------------------------
__global__ __launch_bounds__(256) void gi_render_kernel(
    const float4* __restrict__ gA,
    const float2* __restrict__ gB,
    const v16h*   __restrict__ amatG,
    float*        __restrict__ out)
{
    __shared__ float4 pA[NGAUSS];   // 32 KB: nhA, nB, nhC, cx
    __shared__ float2 pB[NGAUSS];   // 16 KB: cy, opac

    int tid = threadIdx.x;
    for (int i = tid; i < NGAUSS; i += 256) {
        pA[i] = gA[i];
        pB[i] = gB[i];
    }
    __syncthreads();

    int wave = tid >> 5;
    int lane = tid & 31;
    int pixBase = blockIdx.x * 256 + wave * 32;

    // B-matrix column for this lane (both lane halves map to columns 0..15)
    int p0  = pixBase + (lane & 15);
    float fx0 = (float)(p0 & (IMG_W - 1)) + 0.5f;
    float fx1 = fx0 + 16.0f;               // second pixel group, same row
    float fy  = (float)(p0 >> 8) + 0.5f;
    // B-matrix K range: lanes 0-15 -> K={0..7,16..23}, lanes 16-31 -> K={8..15,24..31}
    int khalf = (lane & 16) >> 1;

    v8f acc0 = {};
    v8f acc1 = {};

    for (int kb = 0; kb < NBLK; ++kb) {
        v16h a = amatG[kb * 32 + lane];    // pre-swizzled colors (32B load)
        int kbaseB = kb * 32 + khalf;
        v16h b0 = {};
        v16h b1 = {};
#pragma unroll
        for (int e = 0; e < 16; ++e) {
            int k = kbaseB + e + (e & 8);
            float4 pa = pA[k];             // ds_load_b128 (broadcast)
            float2 pb = pB[k];             // ds_load_b64  (broadcast)
            float dy  = fy - pb.x;
            float dyC = (pa.z * dy) * dy;                       // nhC*dy^2
            float Bdy = pa.y * dy;                              // nB*dy
            float dx0 = fx0 - pa.w;
            float dx1 = fx1 - pa.w;
            // s = -(0.5*A*dx^2 + 0.5*C*dy^2 + B*dx*dy) * log2(e)
            float s0  = __builtin_fmaf(dx0, __builtin_fmaf(pa.x, dx0, Bdy), dyC);
            float s1  = __builtin_fmaf(dx1, __builtin_fmaf(pa.x, dx1, Bdy), dyC);
            b0[e] = (_Float16)fminf(0.99f, pb.y * __builtin_amdgcn_exp2f(s0));
            b1[e] = (_Float16)fminf(0.99f, pb.y * __builtin_amdgcn_exp2f(s1));
        }
        // D = colors^T (16x32) x alpha (32x16) + C
        acc0 = __builtin_amdgcn_wmma_f32_16x16x32_f16(
                   false, a, false, b0, (short)0, acc0, false, false);
        acc1 = __builtin_amdgcn_wmma_f32_16x16x32_f16(
                   false, a, false, b1, (short)0, acc1, false, false);
    }

    // D rows 0..2 (channels) live in acc elements 0..2 of lanes 0..15,
    // column N = lane. Lanes 16..31 hold rows 8..15 (unused).
    if (lane < 16) {
#pragma unroll
        for (int c = 0; c < 3; ++c) {
            float v0 = fminf(fmaxf(acc0[c], 0.0f), 1.0f);
            float v1 = fminf(fmaxf(acc1[c], 0.0f), 1.0f);
            out[c * NPIX + p0]      = v0;
            out[c * NPIX + p0 + 16] = v1;
        }
    }
}

// ---------------------------------------------------------------------------
extern "C" void kernel_launch(void* const* d_in, const int* in_sizes, int n_in,
                              void* d_out, int out_size, void* d_ws, size_t ws_size,
                              hipStream_t stream)
{
    const float* xyz  = (const float*)d_in[0];   // (3, 2048, 2)
    const float* chol = (const float*)d_in[1];   // (3, 2048, 3)
    const float* opac = (const float*)d_in[2];   // (2048, 1)
    const float* feat = (const float*)d_in[3];   // (2048, 3)
    const int*   fidx = (const int*)d_in[4];     // scalar

    char* ws = (char*)d_ws;
    float4*   packA = (float4*)ws;                        // 32 KB
    float2*   packB = (float2*)(ws + NGAUSS * 16);        // 16 KB @ 32768
    _Float16* amat  = (_Float16*)(ws + NGAUSS * 24);      // 64 KB @ 49152

    gi_params_kernel<<<NGAUSS / 256, 256, 0, stream>>>(xyz, chol, opac, fidx,
                                                       packA, packB);
    gi_amat_kernel<<<NGAUSS / 256, 256, 0, stream>>>(feat, amat);
    gi_render_kernel<<<NPIX / 256, 256, 0, stream>>>(packA, packB,
                                                     (const v16h*)amat,
                                                     (float*)d_out);
}
